// Tacotron_13812614824236
// MI455X (gfx1250) — matrix operand order repaired and checked by reading, not compiled, not run on hardware
//
#include <hip/hip_runtime.h>
#include <math.h>

typedef __attribute__((ext_vector_type(16))) __bf16 v16bf;
typedef __attribute__((ext_vector_type(8)))  float  v8f;

#define TENC 128
#define TDEC 200
#define NB   32

__device__ __forceinline__ float sigf(float x){ return 1.0f/(1.0f+__expf(-x)); }

// ---------------------------------------------------------------------------
// f32 -> bf16 weight conversion with K padding: dst is [N, Kpad], src [N, Kreal]
// ---------------------------------------------------------------------------
__global__ void cvt_w_k(const float* __restrict__ src, __bf16* __restrict__ dst,
                        int N, int Kreal, int Kpad)
{
  int i = blockIdx.x * blockDim.x + threadIdx.x;
  if (i >= N * Kpad) return;
  int n = i / Kpad, k = i - n * Kpad;
  dst[i] = (k < Kreal) ? (__bf16)src[(size_t)n * Kreal + k] : (__bf16)0.0f;
}

// ---------------------------------------------------------------------------
// WMMA tile loaders
// B (= W^T tile): lane<16 holds K=k0+{0..7,16..23} of column `col`,
// lane>=16 holds K=k0+{8..15,24..31}.
// ---------------------------------------------------------------------------
__device__ __forceinline__ v16bf load_b_tile(const __bf16* __restrict__ wrow,
                                             int k0, int hi)
{
  v16bf b;
  const __bf16* wp = wrow + k0 + hi * 8;
  __builtin_prefetch((const void*)(wp + 64), 0, 1);
#pragma unroll
  for (int e = 0; e < 16; ++e) b[e] = wp[e < 8 ? e : e + 8];
  return b;
}

// A 16x32 bf16 layout: lane<16 -> K {0..7,16..23}, lane>=16 -> {8..15,24..31}
template <int MODE>
__device__ __forceinline__ v16bf load_a_tile(const float* __restrict__ abase,
                                             const float* __restrict__ A,
                                             int k0, int koff, int Kreal,
                                             int KW, int pad, int Tc, int Cc,
                                             int bi, int tt)
{
  v16bf a;
  if constexpr (MODE != 1) {
    const float* ap = abase + k0 + koff;
#pragma unroll
    for (int j = 0; j < 8; ++j) a[j]     = (__bf16)ap[j];
#pragma unroll
    for (int j = 0; j < 8; ++j) a[8 + j] = (__bf16)ap[16 + j];
  } else {
#pragma unroll
    for (int j = 0; j < 16; ++j) {
      const int k = k0 + koff + (j < 8 ? j : j + 8);
      float v = 0.0f;
      if (k < Kreal) {
        const int c  = k / KW;
        const int dk = k - c * KW;
        const int st = tt + dk - pad;
        if (st >= 0 && st < Tc) v = A[((size_t)bi * Tc + st) * Cc + c];
      }
      a[j] = (__bf16)v;
    }
  }
  return a;
}

// ---------------------------------------------------------------------------
// Generic WMMA bf16 GEMM:  C[M,N] = epilogue( A[M,K] * W[N,K]^T (+bias)(+betaC) )
// MODE 0: A[row*lda + k]            (plain, fp32 activations)
// MODE 1: im2col conv1d channel-last: row=(b*Tc+t), k=c*KW+dk
// MODE 2: gather rows: A[gidx[row]*lda + k]   (embedding lookup)
// epilogue: act(0 none,1 tanh,2 sigmoid,3 relu);
//           if scale: v = v*scale[n]*smul + shift[n]; if resid: v += resid
// One wave computes one 16x16 tile; K unrolled by 2 with dual accumulators.
// ---------------------------------------------------------------------------
template <int MODE>
__global__ __launch_bounds__(128)
void gemm_wmma(const float* __restrict__ A, const __bf16* __restrict__ W,
               const float* __restrict__ bias, const float* __restrict__ scale,
               const float* __restrict__ shift, const float* __restrict__ resid,
               float* __restrict__ C,
               int M, int N, int Kreal, int Kpad, int lda, int ldw, int wkoff,
               int ldc, int beta, int act, float smul,
               int Tc, int Cc, int KW, int pad,
               const int* __restrict__ gidx)
{
  const int lane = threadIdx.x & 31;
  const int wid  = blockIdx.x * (blockDim.x >> 5) + (threadIdx.x >> 5);
  const int mt = M >> 4, nt = N >> 4;
  if (wid >= mt * nt) return;
  const int mTile = wid % mt;
  const int nTile = wid / mt;
  const int l15 = lane & 15;
  const int hi  = lane >> 4;          // which K-half this lane holds
  const int arow = mTile * 16 + l15;  // A row for this lane
  const int col  = nTile * 16 + l15;  // W row / C column for this lane
  const int koff = hi * 8;

  v8f acc0, acc1;
  {
    const float bv = bias ? bias[col] : 0.0f;
#pragma unroll
    for (int r = 0; r < 8; ++r) {
      float v = bv;
      if (beta) v += C[(size_t)(mTile * 16 + r + hi * 8) * ldc + col];
      acc0[r] = v;
      acc1[r] = 0.0f;
    }
  }

  const __bf16* wrow = W + (size_t)col * ldw + wkoff;
  const float* abase = A;
  int bi = 0, tt = 0;
  if constexpr (MODE == 0) {
    abase = A + (size_t)arow * lda;
  } else if constexpr (MODE == 2) {
    abase = A + (size_t)gidx[arow] * lda;
  } else {
    bi = arow / Tc; tt = arow - bi * Tc;
  }

  const int kpair = Kpad & ~63;
  int k0 = 0;
  for (; k0 < kpair; k0 += 64) {
    v16bf a0 = load_a_tile<MODE>(abase, A, k0, koff, Kreal, KW, pad, Tc, Cc, bi, tt);
    v16bf b0 = load_b_tile(wrow, k0, hi);
    v16bf a1 = load_a_tile<MODE>(abase, A, k0 + 32, koff, Kreal, KW, pad, Tc, Cc, bi, tt);
    v16bf b1 = load_b_tile(wrow, k0 + 32, hi);
    acc0 = __builtin_amdgcn_wmma_f32_16x16x32_bf16(false, a0, false, b0,
                                                   (short)0, acc0, false, false);
    acc1 = __builtin_amdgcn_wmma_f32_16x16x32_bf16(false, a1, false, b1,
                                                   (short)0, acc1, false, false);
  }
  if (k0 < Kpad) {  // odd 32-chunk tail
    v16bf a0 = load_a_tile<MODE>(abase, A, k0, koff, Kreal, KW, pad, Tc, Cc, bi, tt);
    v16bf b0 = load_b_tile(wrow, k0, hi);
    acc0 = __builtin_amdgcn_wmma_f32_16x16x32_bf16(false, a0, false, b0,
                                                   (short)0, acc0, false, false);
  }

#pragma unroll
  for (int r = 0; r < 8; ++r) {
    const size_t orow = (size_t)(mTile * 16 + r + hi * 8);
    float v = acc0[r] + acc1[r];
    if      (act == 1) v = tanhf(v);
    else if (act == 2) v = sigf(v);
    else if (act == 3) v = fmaxf(v, 0.0f);
    if (scale) v = v * (scale[col] * smul) + (shift ? shift[col] : 0.0f);
    if (resid) v += resid[orow * ldc + col];
    C[orow * ldc + col] = v;
  }
}

// ---------------------------------------------------------------------------
// Elementwise / cell kernels
// ---------------------------------------------------------------------------
__global__ void maxpool_k(const float* __restrict__ cb, float* __restrict__ mp,
                          int total)  // [4096 rows, 2048 ch], k=2 s=1 p=1 maxpool
{
  int i = blockIdx.x * blockDim.x + threadIdx.x;
  if (i >= total) return;
  int row = i >> 11;            // /2048
  int t = row & (TENC - 1);
  float v = cb[i];
  if (t > 0) v = fmaxf(v, cb[i - 2048]);
  mp[i] = v;
}

__global__ void hw_comb_k(const float* __restrict__ x1r, const float* __restrict__ g,
                          float* __restrict__ x, int total)
{
  int i = blockIdx.x * blockDim.x + threadIdx.x;
  if (i >= total) return;
  float gv = g[i];
  x[i] = gv * x1r[i] + (1.0f - gv) * x[i];
}

__global__ void gru_cell_k(const float* __restrict__ gi, int gis,
                           const float* __restrict__ gh,
                           float* __restrict__ h,
                           float* __restrict__ out2, int o2s,
                           int Bn, int H)
{
  int i = blockIdx.x * blockDim.x + threadIdx.x;
  if (i >= Bn * H) return;
  int b = i / H, j = i - b * H;
  const float* gib = gi + (size_t)b * gis;
  const float* ghb = gh + (size_t)b * 3 * H;
  float r = sigf(gib[j] + ghb[j]);
  float z = sigf(gib[H + j] + ghb[H + j]);
  float n = tanhf(gib[2 * H + j] + r * ghb[2 * H + j]);
  float hn = (1.0f - z) * n + z * h[(size_t)b * H + j];
  h[(size_t)b * H + j] = hn;
  if (out2) out2[(size_t)b * o2s + j] = hn;
}

__global__ void lstm_cell_k(const float* __restrict__ g,
                            float* __restrict__ h, float* __restrict__ c,
                            const float* __restrict__ xin, float* __restrict__ xout,
                            int Bn, int H)
{
  int i = blockIdx.x * blockDim.x + threadIdx.x;
  if (i >= Bn * H) return;
  int b = i / H, j = i - b * H;
  const float* gb = g + (size_t)b * 4 * H;
  float c2 = sigf(gb[H + j]) * c[i] + sigf(gb[j]) * tanhf(gb[2 * H + j]);
  float hn = sigf(gb[3 * H + j]) * tanhf(c2);
  c[i] = c2;
  h[i] = hn;
  if (xout) xout[i] = (xin ? xin[i] : 0.0f) + hn;
}

// u[b,t'] = sum_j tanh(pq[b,j] + encp[b,t',j] + ploc2[b,t',j]) * v[j]
__global__ void attn_u_k(const float* __restrict__ pq, const float* __restrict__ encp,
                         const float* __restrict__ ploc2, const float* __restrict__ vw,
                         float* __restrict__ u)
{
  int w = threadIdx.x >> 5, lane = threadIdx.x & 31;
  int row = blockIdx.x * 8 + w;
  if (row >= NB * TENC) return;
  int b = row >> 7;
  float s = 0.0f;
#pragma unroll
  for (int q = 0; q < 8; ++q) {
    int j = lane + q * 32;
    s += tanhf(pq[b * 256 + j] + encp[(size_t)row * 256 + j] +
               ploc2[(size_t)row * 256 + j]) * vw[j];
  }
  for (int o = 16; o > 0; o >>= 1) s += __shfl_down(s, o, 32);
  if (lane == 0) u[row] = s;
}

// scores = sigmoid(u)/sum; cum += scores; attn = scores; write output slice
__global__ void score_k(const float* __restrict__ u, float* __restrict__ loc,
                        float* __restrict__ outs, int tstep)
{
  int b = blockIdx.x, t = threadIdx.x;   // 128 threads
  __shared__ float red[TENC];
  float su = sigf(u[b * TENC + t]);
  red[t] = su;
  __syncthreads();
  for (int s = 64; s > 0; s >>= 1) { if (t < s) red[t] += red[t + s]; __syncthreads(); }
  float sc = su / red[0];
  loc[(b * TENC + t) * 2 + 0] += sc;
  loc[(b * TENC + t) * 2 + 1]  = sc;
  outs[((size_t)b * TDEC + tstep) * TENC + t] = sc;
}

// ctx[b,d] = sum_t scores[b,t] * enc[b,t,d]
__global__ void ctx_k(const float* __restrict__ loc, const float* __restrict__ enc,
                      float* __restrict__ ctx)
{
  int b = blockIdx.x, d = threadIdx.x;   // 256 threads
  float s = 0.0f;
  for (int t = 0; t < TENC; ++t)
    s += loc[(b * TENC + t) * 2 + 1] * enc[((size_t)b * TENC + t) * 256 + d];
  ctx[b * 256 + d] = s;
}

__global__ void concat_rx_k(const float* __restrict__ mel, const float* __restrict__ x,
                            float* __restrict__ rx, int tstep)
{
  int i = blockIdx.x * blockDim.x + threadIdx.x;
  if (i >= NB * 672) return;
  int b = i / 672, j = i - b * 672;
  rx[i] = (j < 160) ? mel[((size_t)b * TDEC + tstep) * 160 + j]
                    : x[b * 512 + (j - 160)];
}

__global__ void stop_k(const float* __restrict__ rx, const float* __restrict__ sw,
                       const float* __restrict__ sb, float* __restrict__ outstop,
                       int tstep)
{
  int b = threadIdx.x >> 5, lane = threadIdx.x & 31;   // 1024 threads = 32 waves
  float s = 0.0f;
#pragma unroll
  for (int q = 0; q < 21; ++q) { int j = lane + q * 32; s += sw[j] * rx[b * 672 + j]; }
  for (int o = 16; o > 0; o >>= 1) s += __shfl_down(s, o, 32);
  if (lane == 0) outstop[b * TDEC + tstep] = sigf(s + sb[0]);
}

// ---------------------------------------------------------------------------
// Host orchestration
// ---------------------------------------------------------------------------
static inline int cdiv(int a, int b) { return (a + b - 1) / b; }

static void gemm(hipStream_t st, const float* A, const __bf16* W, const float* bias,
                 const float* scale, const float* shift, const float* resid, float* C,
                 int M, int N, int Kreal, int Kpad, int lda, int ldw, int wkoff,
                 int ldc, int beta, int act, float smul,
                 int mode, int Tc, int Cc, int KW, int pad, const int* gidx)
{
  int tiles = (M / 16) * (N / 16);
  dim3 g(cdiv(tiles, 4)), bdim(128);
  if (mode == 0)
    gemm_wmma<0><<<g, bdim, 0, st>>>(A, W, bias, scale, shift, resid, C,
        M, N, Kreal, Kpad, lda, ldw, wkoff, ldc, beta, act, smul, Tc, Cc, KW, pad, gidx);
  else if (mode == 1)
    gemm_wmma<1><<<g, bdim, 0, st>>>(A, W, bias, scale, shift, resid, C,
        M, N, Kreal, Kpad, lda, ldw, wkoff, ldc, beta, act, smul, Tc, Cc, KW, pad, gidx);
  else
    gemm_wmma<2><<<g, bdim, 0, st>>>(A, W, bias, scale, shift, resid, C,
        M, N, Kreal, Kpad, lda, ldw, wkoff, ldc, beta, act, smul, Tc, Cc, KW, pad, gidx);
}

extern "C" void kernel_launch(void* const* d_in, const int* in_sizes, int n_in,
                              void* d_out, int out_size, void* d_ws, size_t ws_size,
                              hipStream_t stream)
{
  (void)in_sizes; (void)n_in; (void)out_size; (void)ws_size;
  // ---- inputs: text, mel, then params as pytree leaves (sorted keys) ----
  const int*   text = (const int*)d_in[0];
  const float* mel  = (const float*)d_in[1];
  const float* agru_bhh = (const float*)d_in[2];
  const float* agru_bih = (const float*)d_in[3];
  const float* agru_whh = (const float*)d_in[4];
  const float* agru_wih = (const float*)d_in[5];
  const float* bank_b[16]; const float* bank_g[16]; const float* bank_w[16];
  for (int i = 0; i < 16; ++i) {
    bank_b[i] = (const float*)d_in[6 + i];
    bank_g[i] = (const float*)d_in[22 + i];
    bank_w[i] = (const float*)d_in[38 + i];
  }
  const float* dpre1_b = (const float*)d_in[54];
  const float* dpre1_w = (const float*)d_in[55];
  const float* dpre2_b = (const float*)d_in[56];
  const float* dpre2_w = (const float*)d_in[57];
  const float* embed   = (const float*)d_in[58];
  const float* fc1_b   = (const float*)d_in[59];
  const float* fc1_w   = (const float*)d_in[60];
  const float* fc2_b   = (const float*)d_in[61];
  const float* fc2_w   = (const float*)d_in[62];
  const float* encp_w  = (const float*)d_in[63];
  const float* grub_bhh = (const float*)d_in[64];
  const float* grub_bih = (const float*)d_in[65];
  const float* grub_whh = (const float*)d_in[66];
  const float* grub_wih = (const float*)d_in[67];
  const float* gruf_bhh = (const float*)d_in[68];
  const float* gruf_bih = (const float*)d_in[69];
  const float* gruf_whh = (const float*)d_in[70];
  const float* gruf_wih = (const float*)d_in[71];
  const float* hw_b1 = (const float*)d_in[72];
  const float* hw_b2 = (const float*)d_in[73];
  const float* hw_w1 = (const float*)d_in[74];
  const float* hw_w2 = (const float*)d_in[75];
  const float* lsa_L_b = (const float*)d_in[76];
  const float* lsa_L_w = (const float*)d_in[77];
  const float* lsa_W_b = (const float*)d_in[78];
  const float* lsa_W_w = (const float*)d_in[79];
  const float* lsa_conv = (const float*)d_in[80];
  const float* lsa_v_w  = (const float*)d_in[81];
  const float* l1_bhh = (const float*)d_in[82];
  const float* l1_bih = (const float*)d_in[83];
  const float* l1_whh = (const float*)d_in[84];
  const float* l1_wih = (const float*)d_in[85];
  const float* l2_bhh = (const float*)d_in[86];
  const float* l2_bih = (const float*)d_in[87];
  const float* l2_whh = (const float*)d_in[88];
  const float* l2_wih = (const float*)d_in[89];
  const float* p1_b = (const float*)d_in[90];
  const float* p1_g = (const float*)d_in[91];
  const float* p1_w = (const float*)d_in[92];
  const float* p2_b = (const float*)d_in[93];
  const float* p2_g = (const float*)d_in[94];
  const float* p2_w = (const float*)d_in[95];
  const float* res_bhh = (const float*)d_in[96];
  const float* res_bih = (const float*)d_in[97];
  const float* res_whh = (const float*)d_in[98];
  const float* res_wih = (const float*)d_in[99];
  const float* rnn_b = (const float*)d_in[100];
  const float* rnn_w = (const float*)d_in[101];
  const float* stop_b = (const float*)d_in[102];
  const float* stop_w = (const float*)d_in[103];

  float* out_stops  = (float*)d_out;
  float* out_scores = out_stops + NB * TDEC;

  const float BN_S = 1.0f / sqrtf(1.0f + 1e-5f);

  // ---- workspace layout ----
  size_t off = 0;
  auto alloc = [&](size_t bytes) -> void* {
    off = (off + 255) & ~(size_t)255;
    void* p = (char*)d_ws + off;
    off += bytes;
    return p;
  };
  auto fball = [&](size_t n) -> float* { return (float*)alloc(n * 4); };
  auto bball = [&](size_t n) -> __bf16* { return (__bf16*)alloc(n * 2); };

  // bf16 weights (padded K layout)
  __bf16* wb_fc1  = bball(256 * 256);
  __bf16* wb_fc2  = bball(128 * 256);
  __bf16* wb_bank[16];
  for (int i = 0; i < 16; ++i) wb_bank[i] = bball((size_t)128 * 128 * (i + 1));
  __bf16* wb_p1   = bball((size_t)128 * 6144);
  __bf16* wb_p2   = bball(128 * 384);
  __bf16* wb_hw1  = bball(512 * 128);
  __bf16* wb_hw2  = bball(512 * 128);
  __bf16* wb_gfi  = bball(384 * 128);
  __bf16* wb_gfh  = bball(384 * 128);
  __bf16* wb_gbi  = bball(384 * 128);
  __bf16* wb_gbh  = bball(384 * 128);
  __bf16* wb_ep   = bball(256 * 256);
  __bf16* wb_dp1  = bball(256 * 160);
  __bf16* wb_dp2  = bball(128 * 256);
  __bf16* wb_lc   = bball(32 * 64);
  __bf16* wb_lL   = bball(256 * 32);
  __bf16* wb_lW   = bball(256 * 256);
  __bf16* wb_awi  = bball(768 * 384);
  __bf16* wb_awh  = bball(768 * 256);
  __bf16* wb_rnn  = bball(512 * 512);
  __bf16* wb_l1i  = bball((size_t)2048 * 512);
  __bf16* wb_l1h  = bball((size_t)2048 * 512);
  __bf16* wb_l2i  = bball((size_t)2048 * 512);
  __bf16* wb_l2h  = bball((size_t)2048 * 512);
  __bf16* wb_ri   = bball((size_t)10752 * 672);
  __bf16* wb_rh   = bball((size_t)10752 * 672);

  // fp32 activations
  const int MRows = NB * TENC;  // 4096
  float* h1buf  = fball((size_t)MRows * 256);
  float* xe     = fball((size_t)MRows * 128);
  float* cb     = fball((size_t)MRows * 2048);
  float* mp     = fball((size_t)MRows * 2048);
  float* ph     = fball((size_t)MRows * 128);
  float* xenc   = fball((size_t)MRows * 128);
  float* hx1    = fball((size_t)MRows * 128);
  float* hg     = fball((size_t)MRows * 128);
  float* gif    = fball((size_t)MRows * 384);
  float* gib    = fball((size_t)MRows * 384);
  float* ghbuf  = fball(NB * 384);
  float* enc    = fball((size_t)MRows * 256);
  float* encp   = fball((size_t)MRows * 256);
  float* dt1    = fball((size_t)NB * TDEC * 256);
  float* preall = fball((size_t)NB * TDEC * 128);
  float* gi_d   = fball(NB * 768);
  float* gh_d   = fball(NB * 768);
  float* pq     = fball(NB * 256);
  float* plocF  = fball((size_t)MRows * 32);
  float* ploc2  = fball((size_t)MRows * 256);
  float* ubuf   = fball(MRows);
  float* xbuf   = fball(NB * 512);
  float* xs1    = fball(NB * 512);
  float* xs2    = fball(NB * 512);
  float* gbuf   = fball(NB * 2688);
  float* rx     = fball(NB * 672);

  // zero-initialized state (contiguous block, one memset)
  size_t zn = 0;
  float* zbase = fball(8192 + 8192 + 4 * 16384 + 2 * 86016 + 4096 + 4096 + 8192);
  float* attn_h = zbase;            zn += 8192;   // 32*256
  float* ctxb   = zbase + zn;       zn += 8192;   // 32*256
  float* h1s    = zbase + zn;       zn += 16384;  // 32*512
  float* c1s    = zbase + zn;       zn += 16384;
  float* h2s    = zbase + zn;       zn += 16384;
  float* c2s    = zbase + zn;       zn += 16384;
  float* rhs    = zbase + zn;       zn += 86016;  // 4*32*672
  float* rcs    = zbase + zn;       zn += 86016;
  float* hf     = zbase + zn;       zn += 4096;   // 32*128
  float* hb     = zbase + zn;       zn += 4096;
  float* locbuf = zbase + zn;       zn += 8192;   // [B,T,2] cum/attn
  hipMemsetAsync(zbase, 0, zn * 4, stream);

  // ---- weight conversion (per call; deterministic) ----
  auto cvt = [&](const float* s, __bf16* d, int N, int Kr, int Kp) {
    cvt_w_k<<<cdiv(N * Kp, 256), 256, 0, stream>>>(s, d, N, Kr, Kp);
  };
  cvt(fc1_w, wb_fc1, 256, 256, 256);
  cvt(fc2_w, wb_fc2, 128, 256, 256);
  for (int i = 0; i < 16; ++i) cvt(bank_w[i], wb_bank[i], 128, 128 * (i + 1), 128 * (i + 1));
  cvt(p1_w, wb_p1, 128, 6144, 6144);
  cvt(p2_w, wb_p2, 128, 384, 384);
  cvt(hw_w1, wb_hw1, 512, 128, 128);
  cvt(hw_w2, wb_hw2, 512, 128, 128);
  cvt(gruf_wih, wb_gfi, 384, 128, 128);  cvt(gruf_whh, wb_gfh, 384, 128, 128);
  cvt(grub_wih, wb_gbi, 384, 128, 128);  cvt(grub_whh, wb_gbh, 384, 128, 128);
  cvt(encp_w, wb_ep, 256, 256, 256);
  cvt(dpre1_w, wb_dp1, 256, 160, 160);
  cvt(dpre2_w, wb_dp2, 128, 256, 256);
  cvt(lsa_conv, wb_lc, 32, 62, 64);
  cvt(lsa_L_w, wb_lL, 256, 32, 32);
  cvt(lsa_W_w, wb_lW, 256, 256, 256);
  cvt(agru_wih, wb_awi, 768, 384, 384);
  cvt(agru_whh, wb_awh, 768, 256, 256);
  cvt(rnn_w, wb_rnn, 512, 512, 512);
  cvt(l1_wih, wb_l1i, 2048, 512, 512);  cvt(l1_whh, wb_l1h, 2048, 512, 512);
  cvt(l2_wih, wb_l2i, 2048, 512, 512);  cvt(l2_whh, wb_l2h, 2048, 512, 512);
  cvt(res_wih, wb_ri, 10752, 672, 672);
  cvt(res_whh, wb_rh, 10752, 672, 672);

  // ================= ENCODER =================
  // embedding gather + prenet fc1 (tanh), fc2 (tanh)
  gemm(stream, embed, wb_fc1, fc1_b, 0, 0, 0, h1buf,
       MRows, 256, 256, 256, 256, 256, 0, 256, 0, 1, 1.0f, 2, 0, 0, 0, 0, text);
  gemm(stream, h1buf, wb_fc2, fc2_b, 0, 0, 0, xe,
       MRows, 128, 256, 256, 256, 256, 0, 128, 0, 1, 1.0f, 0, 0, 0, 0, 0, 0);

  // conv banks -> bn(relu(conv)) fused, concat channel-last into cb[4096,2048]
  for (int i = 0; i < 16; ++i) {
    int k = i + 1;
    gemm(stream, xe, wb_bank[i], 0, bank_g[i], bank_b[i], 0, cb + i * 128,
         MRows, 128, 128 * k, 128 * k, 0, 128 * k, 0, 2048, 0, 3, BN_S,
         1, TENC, 128, k, k / 2, 0);
  }
  maxpool_k<<<cdiv(MRows * 2048, 256), 256, 0, stream>>>(cb, mp, MRows * 2048);

  // projections (+BN), residual add fused into proj2 epilogue
  gemm(stream, mp, wb_p1, 0, p1_g, p1_b, 0, ph,
       MRows, 128, 6144, 6144, 0, 6144, 0, 128, 0, 3, BN_S, 1, TENC, 2048, 3, 1, 0);
  gemm(stream, ph, wb_p2, 0, p2_g, p2_b, xe, xenc,
       MRows, 128, 384, 384, 0, 384, 0, 128, 0, 0, BN_S, 1, TENC, 128, 3, 1, 0);

  // highway x4
  for (int i = 0; i < 4; ++i) {
    gemm(stream, xenc, wb_hw1 + (size_t)i * 128 * 128, hw_b1 + i * 128, 0, 0, 0, hx1,
         MRows, 128, 128, 128, 128, 128, 0, 128, 0, 3, 1.0f, 0, 0, 0, 0, 0, 0);
    gemm(stream, xenc, wb_hw2 + (size_t)i * 128 * 128, hw_b2 + i * 128, 0, 0, 0, hg,
         MRows, 128, 128, 128, 128, 128, 0, 128, 0, 2, 1.0f, 0, 0, 0, 0, 0, 0);
    hw_comb_k<<<cdiv(MRows * 128, 256), 256, 0, stream>>>(hx1, hg, xenc, MRows * 128);
  }

  // bi-GRU: input-side gates for all timesteps in one shot
  gemm(stream, xenc, wb_gfi, gruf_bih, 0, 0, 0, gif,
       MRows, 384, 128, 128, 128, 128, 0, 384, 0, 0, 1.0f, 0, 0, 0, 0, 0, 0);
  gemm(stream, xenc, wb_gbi, grub_bih, 0, 0, 0, gib,
       MRows, 384, 128, 128, 128, 128, 0, 384, 0, 0, 1.0f, 0, 0, 0, 0, 0, 0);
  for (int t = 0; t < TENC; ++t) {    // forward
    gemm(stream, hf, wb_gfh, gruf_bhh, 0, 0, 0, ghbuf,
         NB, 384, 128, 128, 128, 128, 0, 384, 0, 0, 1.0f, 0, 0, 0, 0, 0, 0);
    gru_cell_k<<<cdiv(NB * 128, 256), 256, 0, stream>>>(
        gif + (size_t)t * 384, TENC * 384, ghbuf, hf,
        enc + (size_t)t * 256, TENC * 256, NB, 128);
  }
  for (int s = 0; s < TENC; ++s) {    // backward
    int t = TENC - 1 - s;
    gemm(stream, hb, wb_gbh, grub_bhh, 0, 0, 0, ghbuf,
         NB, 384, 128, 128, 128, 128, 0, 384, 0, 0, 1.0f, 0, 0, 0, 0, 0, 0);
    gru_cell_k<<<cdiv(NB * 128, 256), 256, 0, stream>>>(
        gib + (size_t)t * 384, TENC * 384, ghbuf, hb,
        enc + (size_t)t * 256 + 128, TENC * 256, NB, 128);
  }

  // enc_proj
  gemm(stream, enc, wb_ep, 0, 0, 0, 0, encp,
       MRows, 256, 256, 256, 256, 256, 0, 256, 0, 0, 1.0f, 0, 0, 0, 0, 0, 0);

  // decoder prenet for all timesteps
  gemm(stream, mel, wb_dp1, dpre1_b, 0, 0, 0, dt1,
       NB * TDEC, 256, 160, 160, 160, 160, 0, 256, 0, 1, 1.0f, 0, 0, 0, 0, 0, 0);
  gemm(stream, dt1, wb_dp2, dpre2_b, 0, 0, 0, preall,
       NB * TDEC, 128, 256, 256, 256, 256, 0, 128, 0, 1, 1.0f, 0, 0, 0, 0, 0, 0);

  // ================= DECODER =================
  for (int t = 0; t < TDEC; ++t) {
    // attention GRU: gi = [ctx, pre_t] @ Wih^T + bih (K split via wkoff)
    gemm(stream, ctxb, wb_awi, agru_bih, 0, 0, 0, gi_d,
         NB, 768, 256, 256, 256, 384, 0, 768, 0, 0, 1.0f, 0, 0, 0, 0, 0, 0);
    gemm(stream, preall + (size_t)t * 128, wb_awi, 0, 0, 0, 0, gi_d,
         NB, 768, 128, 128, TDEC * 128, 384, 256, 768, 1, 0, 1.0f, 0, 0, 0, 0, 0, 0);
    gemm(stream, attn_h, wb_awh, agru_bhh, 0, 0, 0, gh_d,
         NB, 768, 256, 256, 256, 256, 0, 768, 0, 0, 1.0f, 0, 0, 0, 0, 0, 0);
    gru_cell_k<<<cdiv(NB * 256, 256), 256, 0, stream>>>(
        gi_d, 768, gh_d, attn_h, (float*)0, 0, NB, 256);

    // LSA attention
    gemm(stream, attn_h, wb_lW, lsa_W_b, 0, 0, 0, pq,
         NB, 256, 256, 256, 256, 256, 0, 256, 0, 0, 1.0f, 0, 0, 0, 0, 0, 0);
    gemm(stream, locbuf, wb_lc, 0, 0, 0, 0, plocF,          // conv31 over [B,2,T]
         MRows, 32, 62, 64, 0, 64, 0, 32, 0, 0, 1.0f, 1, TENC, 2, 31, 15, 0);
    gemm(stream, plocF, wb_lL, lsa_L_b, 0, 0, 0, ploc2,
         MRows, 256, 32, 32, 32, 32, 0, 256, 0, 0, 1.0f, 0, 0, 0, 0, 0, 0);
    attn_u_k<<<cdiv(MRows, 8), 256, 0, stream>>>(pq, encp, ploc2, lsa_v_w, ubuf);
    score_k<<<NB, TENC, 0, stream>>>(ubuf, locbuf, out_scores, t);
    ctx_k<<<NB, 256, 0, stream>>>(locbuf, enc, ctxb);

    // rnn_in = [ctx, attn_h] @ W^T + b
    gemm(stream, ctxb, wb_rnn, rnn_b, 0, 0, 0, xbuf,
         NB, 512, 256, 256, 256, 512, 0, 512, 0, 0, 1.0f, 0, 0, 0, 0, 0, 0);
    gemm(stream, attn_h, wb_rnn, 0, 0, 0, 0, xbuf,
         NB, 512, 256, 256, 256, 512, 256, 512, 1, 0, 1.0f, 0, 0, 0, 0, 0, 0);

    // LSTM1 (+residual into xs1), LSTM2 (+residual into xs2)
    gemm(stream, xbuf, wb_l1i, l1_bih, 0, 0, 0, gbuf,
         NB, 2048, 512, 512, 512, 512, 0, 2048, 0, 0, 1.0f, 0, 0, 0, 0, 0, 0);
    gemm(stream, h1s, wb_l1h, l1_bhh, 0, 0, 0, gbuf,
         NB, 2048, 512, 512, 512, 512, 0, 2048, 1, 0, 1.0f, 0, 0, 0, 0, 0, 0);
    lstm_cell_k<<<cdiv(NB * 512, 256), 256, 0, stream>>>(gbuf, h1s, c1s, xbuf, xs1, NB, 512);
    gemm(stream, xs1, wb_l2i, l2_bih, 0, 0, 0, gbuf,
         NB, 2048, 512, 512, 512, 512, 0, 2048, 0, 0, 1.0f, 0, 0, 0, 0, 0, 0);
    gemm(stream, h2s, wb_l2h, l2_bhh, 0, 0, 0, gbuf,
         NB, 2048, 512, 512, 512, 512, 0, 2048, 1, 0, 1.0f, 0, 0, 0, 0, 0, 0);
    lstm_cell_k<<<cdiv(NB * 512, 256), 256, 0, stream>>>(gbuf, h2s, c2s, xs1, xs2, NB, 512);

    // residual LSTM stack on rx = [mel_t, x]
    concat_rx_k<<<cdiv(NB * 672, 256), 256, 0, stream>>>(mel, xs2, rx, t);
    for (int i = 0; i < 4; ++i) {
      gemm(stream, rx, wb_ri + (size_t)i * 2688 * 672, res_bih + i * 2688, 0, 0, 0, gbuf,
           NB, 2688, 672, 672, 672, 672, 0, 2688, 0, 0, 1.0f, 0, 0, 0, 0, 0, 0);
      gemm(stream, rhs + (size_t)i * NB * 672, wb_rh + (size_t)i * 2688 * 672,
           res_bhh + i * 2688, 0, 0, 0, gbuf,
           NB, 2688, 672, 672, 672, 672, 0, 2688, 1, 0, 1.0f, 0, 0, 0, 0, 0, 0);
      lstm_cell_k<<<cdiv(NB * 672, 256), 256, 0, stream>>>(
          gbuf, rhs + (size_t)i * NB * 672, rcs + (size_t)i * NB * 672, rx, rx, NB, 672);
    }
    stop_k<<<1, 1024, 0, stream>>>(rx, stop_w, stop_b, out_stops, t);
  }
}
